// SSDLoss_28905129902508
// MI455X (gfx1250) — compile-verified, matrix operand order checked by reading, not run on hardware
//
#include <hip/hip_runtime.h>
#include <math.h>

#define N_DB  8732
#define NCLS  21
#define NPRED 25
#define NGT   16
#define NB    256

typedef __attribute__((ext_vector_type(2)))  float    v2f;
typedef __attribute__((ext_vector_type(8)))  float    v8f;
typedef __attribute__((ext_vector_type(16))) _Float16 v16h;

// ---- monotone float <-> uint key (ascending key == ascending float) ----
__device__ inline unsigned f2key(float f) {
  unsigned u = __float_as_uint(f);
  return (u & 0x80000000u) ? ~u : (u | 0x80000000u);
}
__device__ inline float key2f(unsigned k) {
  unsigned u = (k & 0x80000000u) ? (k ^ 0x80000000u) : ~k;
  return __uint_as_float(u);
}

// ---- block-wide sum of 256 per-thread partials via the CDNA5 matrix core.
// A = 16x4 tile of partials, B = ones(4x16): D[m][n] = sum_k A[m][k] + C,
// exact f32 accumulation. 4 WMMAs fold 256 values; rows combined by one
// shfl_xor across the two lane halves of the wave32 C layout.
__device__ float block_sum(float v, float* red) {
  const int tid = threadIdx.x;
  red[tid] = v;
  __syncthreads();
  if (tid < 32) {                 // whole wave 0 => EXEC all ones for WMMA
    const int lane = tid;
    float s;
#if !defined(__AMDGCN__)
    s = 0.0f;                     // host-pass stub, never executed
#elif __has_builtin(__builtin_amdgcn_wmma_f32_16x16x4_f32)
    v8f acc = {};
    v2f onesB; onesB.x = 1.0f; onesB.y = 1.0f;
#pragma unroll
    for (int c = 0; c < 4; ++c) {
      v2f a; a.x = red[c * 64 + lane]; a.y = red[c * 64 + 32 + lane];
      acc = __builtin_amdgcn_wmma_f32_16x16x4_f32(false, a, false, onesB,
                                                  (short)0, acc, false, false);
    }
    s = acc[0] + acc[1] + acc[2] + acc[3] + acc[4] + acc[5] + acc[6] + acc[7];
#else
    // fallback: codegen-confirmed f16 WMMA (values converted to f16)
    v16h a, onesB;
#pragma unroll
    for (int i = 0; i < 16; ++i) {
      int idx = lane * 16 + i;    // lanes 16..31 carry zero padding
      a[i]     = (idx < 256) ? (_Float16)red[idx] : (_Float16)0.0f;
      onesB[i] = (_Float16)1.0f;
    }
    v8f c0 = {};
    v8f accD = __builtin_amdgcn_wmma_f32_16x16x32_f16(false, a, false, onesB,
                                                      (short)0, c0, false, false);
    s = accD[0] + accD[1] + accD[2] + accD[3] +
        accD[4] + accD[5] + accD[6] + accD[7];
#endif
    s += __shfl_xor(s, 16, 32);   // combine rows 0..7 (lanes<16) + 8..15
    if (lane == 0) red[0] = s;
  }
  __syncthreads();
  float r = red[0];
  __syncthreads();
  return r;
}

// ---- default boxes (compile-time constant geometry) into workspace ----
__global__ __launch_bounds__(256) void dbox_init(float* __restrict__ dbx,
                                                 float* __restrict__ dbl) {
  int i = blockIdx.x * blockDim.x + threadIdx.x;
  if (i >= N_DB) return;
  const int start[7] = {0, 5776, 7942, 8542, 8692, 8728, 8732};
  const int msA[6]   = {38, 19, 10, 5, 3, 1};
  const float s_[7]  = {21.f/300.f, 45.f/300.f, 99.f/300.f, 153.f/300.f,
                        207.f/300.f, 261.f/300.f, 315.f/300.f};
  int k = 0;
  while (k < 5 && i >= start[k + 1]) ++k;
  int ms = msA[k];
  int r = i - start[k];
  int cells = ms * ms;
  int a = r / cells;
  int cell = r - a * cells;
  int y = cell / ms;
  int x = cell - y * ms;
  float skf = s_[k];
  float shf = sqrtf(s_[k] * s_[k + 1]);
  double sk = (double)skf, sh = (double)shf;
  const double R2 = sqrt(2.0), R3 = sqrt(3.0);
  double w, h;
  switch (a) {
    case 0: w = sk;      h = sk;      break;
    case 1: w = sh;      h = sh;      break;
    case 2: w = sk * R2; h = sk / R2; break;
    case 3: w = sk / R2; h = sk * R2; break;
    case 4: w = sk * R3; h = sk / R3; break;
    default:w = sk / R3; h = sk * R3; break;
  }
  float wf = (float)fmin(w, 1.0);
  float hf = (float)fmin(h, 1.0);
  float cx = ((float)x + 0.5f) / (float)ms;
  float cy = ((float)y + 0.5f) / (float)ms;
  dbx[i*4+0] = cx; dbx[i*4+1] = cy; dbx[i*4+2] = wf; dbx[i*4+3] = hf;
  float hw = wf * 0.5f, hh = hf * 0.5f;
  dbl[i*4+0] = cx - hw; dbl[i*4+1] = cy - hh;
  dbl[i*4+2] = cx + hw; dbl[i*4+3] = cy + hh;
}

// ---- main: one workgroup per image ----
__global__ __launch_bounds__(256) void ssd_loss_kernel(
    const float* __restrict__ y_pred, const float* __restrict__ gt_xywh,
    const int* __restrict__ gt_cls, const float* __restrict__ dbx,
    const float* __restrict__ dbl, float* __restrict__ per_img) {
  __shared__ float    s_neg[N_DB];     // background NLL per default box
  __shared__ float    s_red[256];
  __shared__ int      s_redi[256];
  __shared__ unsigned s_hist[256];
  __shared__ float    s_gtx[NGT][4];   // gt xywh
  __shared__ float    s_gtl[NGT][4];   // gt ltrb
  __shared__ float    s_agt[NGT];
  __shared__ int      s_cls[NGT];
  __shared__ int      s_bestdb[NGT];
  __shared__ unsigned s_sel[2];        // [0]=digit, [1]=remaining rank

  const int b   = blockIdx.x;
  const int tid = threadIdx.x;
  const float* yp = y_pred + (size_t)b * NPRED * N_DB;

  if (tid < NGT) {
    const float* g = gt_xywh + ((size_t)b * NGT + tid) * 4;
    float cx = g[0], cy = g[1], w = g[2], h = g[3];
    s_gtx[tid][0] = cx; s_gtx[tid][1] = cy; s_gtx[tid][2] = w; s_gtx[tid][3] = h;
    float l = cx - w * 0.5f, t = cy - h * 0.5f;
    float rr = cx + w * 0.5f, bb = cy + h * 0.5f;
    s_gtl[tid][0] = l; s_gtl[tid][1] = t; s_gtl[tid][2] = rr; s_gtl[tid][3] = bb;
    s_agt[tid] = (rr - l) * (bb - t);
    s_cls[tid] = gt_cls[b * NGT + tid];
  }
  __syncthreads();

  // ---- pass A: per-GT argmax over default boxes (first-index ties) ----
  float bestv[NGT]; int besti[NGT];
#pragma unroll
  for (int g = 0; g < NGT; ++g) { bestv[g] = -1.0f; besti[g] = 0x7FFFFFFF; }
  for (int j = tid; j < N_DB; j += 256) {
    float dl = dbl[j*4+0], dt = dbl[j*4+1], dr = dbl[j*4+2], dbm = dbl[j*4+3];
    float adb = (dr - dl) * (dbm - dt);
#pragma unroll
    for (int g = 0; g < NGT; ++g) {
      float w = fmaxf(fminf(s_gtl[g][2], dr) - fmaxf(s_gtl[g][0], dl), 0.0f);
      float h = fmaxf(fminf(s_gtl[g][3], dbm) - fmaxf(s_gtl[g][1], dt), 0.0f);
      float inter = w * h;
      float iou = inter / (s_agt[g] + adb - inter);
      if (iou > bestv[g]) { bestv[g] = iou; besti[g] = j; }
    }
  }
  for (int g = 0; g < NGT; ++g) {
    s_red[tid] = bestv[g]; s_redi[tid] = besti[g];
    __syncthreads();
    for (int off = 128; off > 0; off >>= 1) {
      if (tid < off) {
        float v2 = s_red[tid + off]; int i2 = s_redi[tid + off];
        float v1 = s_red[tid];       int i1 = s_redi[tid];
        if (v2 > v1 || (v2 == v1 && i2 < i1)) { s_red[tid] = v2; s_redi[tid] = i2; }
      }
      __syncthreads();
    }
    if (tid == 0) s_bestdb[g] = s_redi[0];
    __syncthreads();
  }

  // ---- pass B: per-db matching + loc + softmax NLL (single y_pred sweep) ----
  float accP = 0.0f, accLoc = 0.0f, accCP = 0.0f;
  for (int j = tid; j < N_DB; j += 256) {
    if (j + 256 < N_DB)   // gfx1250 global_prefetch for next logit chunk
      __builtin_prefetch(yp + (size_t)4 * N_DB + j + 256, 0, 0);
    float dl = dbl[j*4+0], dt = dbl[j*4+1], dr = dbl[j*4+2], dbm = dbl[j*4+3];
    float adb = (dr - dl) * (dbm - dt);
    float cmax = -1.0e30f; int bg = 0;
#pragma unroll
    for (int g = 0; g < NGT; ++g) {
      float w = fmaxf(fminf(s_gtl[g][2], dr) - fmaxf(s_gtl[g][0], dl), 0.0f);
      float h = fmaxf(fminf(s_gtl[g][3], dbm) - fmaxf(s_gtl[g][1], dt), 0.0f);
      float inter = w * h;
      float iou = inter / (s_agt[g] + adb - inter);
      float v = (s_bestdb[g] == j) ? 2.0f : iou;
      if (v > cmax) { cmax = v; bg = g; }   // strict > => first-g ties
    }
    bool pos = cmax > 0.5f;

    float l[NCLS]; float m = -1.0e30f;
#pragma unroll
    for (int c = 0; c < NCLS; ++c) {
      l[c] = yp[(size_t)(4 + c) * N_DB + j];
      m = fmaxf(m, l[c]);
    }
    float se = 0.0f;
#pragma unroll
    for (int c = 0; c < NCLS; ++c) se += expf(l[c] - m);
    float lse = m + logf(se);
    float negv = lse - l[0];

    if (pos) {
      accP += 1.0f;
      accCP += lse - l[s_cls[bg]];
      float dx = dbx[j*4+0], dy = dbx[j*4+1], dw = dbx[j*4+2], dh = dbx[j*4+3];
      float hat0 = (s_gtx[bg][0] - dx) / dw;
      float hat1 = (s_gtx[bg][1] - dy) / dh;
      float hat2 = logf(s_gtx[bg][2] / dw);
      float hat3 = logf(s_gtx[bg][3] / dh);
      float d0 = yp[(size_t)0 * N_DB + j] - hat0;
      float d1 = yp[(size_t)1 * N_DB + j] - hat1;
      float d2 = yp[(size_t)2 * N_DB + j] - hat2;
      float d3 = yp[(size_t)3 * N_DB + j] - hat3;
      float a0 = fabsf(d0), a1 = fabsf(d1), a2 = fabsf(d2), a3 = fabsf(d3);
      accLoc += (a0 < 1.0f ? 0.5f * d0 * d0 : a0 - 0.5f)
              + (a1 < 1.0f ? 0.5f * d1 * d1 : a1 - 0.5f)
              + (a2 < 1.0f ? 0.5f * d2 * d2 : a2 - 0.5f)
              + (a3 < 1.0f ? 0.5f * d3 * d3 : a3 - 0.5f);
      negv = -1.0e9f;
    }
    s_neg[j] = negv;
  }
  __syncthreads();

  float P    = block_sum(accP,   s_red);
  float locS = block_sum(accLoc, s_red);
  float cpS  = block_sum(accCP,  s_red);
  int   k    = 3 * (int)(P + 0.5f);

  // ---- exact top-k sum via 4-pass MSB radix select on float keys ----
  if (tid == 0) s_sel[1] = (unsigned)k;
  __syncthreads();
  unsigned prefix = 0;
  for (int shift = 24; shift >= 0; shift -= 8) {
    s_hist[tid] = 0u;
    __syncthreads();
    for (int j = tid; j < N_DB; j += 256) {
      unsigned key = f2key(s_neg[j]);
      bool match = (shift == 24) || (((key ^ prefix) >> (shift + 8)) == 0u);
      if (match) atomicAdd(&s_hist[(key >> shift) & 255u], 1u);
    }
    __syncthreads();
    if (tid == 0) {
      unsigned w = s_sel[1];
      int d;
      for (d = 255; d > 0; --d) {
        unsigned c = s_hist[d];
        if (w <= c) break;
        w -= c;
      }
      s_sel[0] = (unsigned)d;
      s_sel[1] = w;
    }
    __syncthreads();
    prefix |= (s_sel[0] << shift);
  }
  float vth = key2f(prefix);          // exact value of the k-th largest

  float cg = 0.0f, sg = 0.0f;
  for (int j = tid; j < N_DB; j += 256) {
    unsigned key = f2key(s_neg[j]);
    if (key > prefix) { cg += 1.0f; sg += s_neg[j]; }
  }
  float cntGt = block_sum(cg, s_red);
  float sumGt = block_sum(sg, s_red);

  if (tid == 0) {
    float kf = (float)k;
    float conf_neg = (sumGt + (kf - cntGt) * vth) / kf;
    float loc      = locS / (4.0f * P);
    float conf_pos = cpS / P;
    per_img[b] = loc + conf_pos + conf_neg;
  }
}

// ---- batch finalize: sum(i * per_img[i]) / B, fixed order ----
__global__ __launch_bounds__(256) void ssd_finalize(
    const float* __restrict__ per_img, float* __restrict__ out) {
  __shared__ float s_red[256];
  int tid = threadIdx.x;
  float v = per_img[tid] * (float)tid;   // NB == blockDim == 256
  float s = block_sum(v, s_red);
  if (tid == 0) out[0] = s / (float)NB;
}

extern "C" void kernel_launch(void* const* d_in, const int* in_sizes, int n_in,
                              void* d_out, int out_size, void* d_ws, size_t ws_size,
                              hipStream_t stream) {
  (void)in_sizes; (void)n_in; (void)out_size; (void)ws_size;
  const float* y_pred  = (const float*)d_in[0];
  const float* gt_xywh = (const float*)d_in[1];
  const int*   gt_cls  = (const int*)d_in[2];

  // workspace layout: db_xywh[8732*4] | db_ltrb[8732*4] | per_img[256]
  float* dbx     = (float*)d_ws;
  float* dbl     = dbx + (size_t)N_DB * 4;
  float* per_img = dbl + (size_t)N_DB * 4;

  dbox_init   <<<(N_DB + 255) / 256, 256, 0, stream>>>(dbx, dbl);
  ssd_loss_kernel<<<NB, 256, 0, stream>>>(y_pred, gt_xywh, gt_cls,
                                          dbx, dbl, per_img);
  ssd_finalize<<<1, 256, 0, stream>>>(per_img, (float*)d_out);
}
